// MetaLSTMCell_25331717112207
// MI455X (gfx1250) — matrix operand order repaired and checked by reading, not compiled, not run on hardware
//
#include <hip/hip_runtime.h>

typedef __attribute__((ext_vector_type(16))) __bf16 v16bf;
typedef __attribute__((ext_vector_type(8)))  float  v8f;

#define DEVI __device__ __forceinline__

// Problem dims
constexpr int Bn  = 512;
constexpr int Iw  = 1024;
constexpr int Hw  = 1024;
constexpr int HHw = 256;
constexpr int Ew  = 64;

// Tiling: workgroup = 256 threads = 8 waves (wave32), arranged 4(M) x 2(N).
// WG tile = 128(M) x 64(N); wave tile = 32x32 = 2x2 WMMA 16x16 tiles; K step 32 (bf16).
constexpr int TM   = 128;
constexpr int TN   = 64;
constexpr int TK   = 32;
constexpr int ASTR = 40;   // padded bf16 elems per A row (80B, 16B aligned, bank-friendly)
constexpr int BSTR = 136;  // padded bf16 elems per Bt row (272B, 16B aligned)
constexpr int LDSA = TM * ASTR; // 5120 bf16 = 10240 B
constexpr int LDSB = TK * BSTR; // 4352 bf16 =  8704 B

DEVI float sigm(float x) { return 1.0f / (1.0f + __expf(-x)); }

// ---- fragment loads from LDS ----------------------------------------------
// A fragment 16x32 bf16: lanes 0-15 -> row M=lane, K 0..7 then K 16..23;
// lanes 16-31 -> row M=lane-16, K 8..15 then K 24..31.
DEVI v16bf load_afrag(const __bf16* sA, int rowbase, int lane) {
  const int row  = rowbase + (lane & 15);
  const int koff = (lane >> 4) ? 8 : 0;
  const __bf16* p = sA + row * ASTR + koff;
  union { uint4 u[2]; v16bf v; } t;
  t.u[0] = *(const uint4*)(p);        // K = koff .. koff+7
  t.u[1] = *(const uint4*)(p + 16);   // K = 16+koff .. 16+koff+7
  return t.v;
}

// B fragment 32x16 bf16 (B[k][n] = W[n][k], stored transposed in sB[k][n]):
// lanes 0-15 -> col N=lane, K 0..7 then K 16..23; lanes 16-31 -> col N=lane-16, K 8..15 then K 24..31.
DEVI v16bf load_bfrag(const __bf16* sB, int nbase, int lane) {
  const int n    = nbase + (lane & 15);
  const int koff = (lane >> 4) ? 8 : 0;
  v16bf v;
#pragma unroll
  for (int s = 0; s < 8; ++s) {
    v[s]     = sB[(koff + s) * BSTR + n];
    v[8 + s] = sB[(16 + koff + s) * BSTR + n];
  }
  return v;
}

// ---- cooperative staging with fp32 -> bf16 conversion ----------------------
DEVI void stage_tiles(const float* A, long lda, const float* W, long ldw,
                      int m0, int k, int tid, __bf16* sA, __bf16* sB) {
  // A tile: 128 rows x 32 K
#pragma unroll
  for (int p = 0; p < 4; ++p) {
    const int row = p * 32 + (tid >> 3);
    const int kc  = (tid & 7) * 4;
    const float4 v = *(const float4*)(A + (long)(m0 + row) * lda + k + kc);
    __bf16* d = sA + row * ASTR + kc;
    d[0] = (__bf16)v.x; d[1] = (__bf16)v.y; d[2] = (__bf16)v.z; d[3] = (__bf16)v.w;
  }
  // W tile: 64 rows (N) x 32 K, stored transposed into sB[k][n]
#pragma unroll
  for (int p = 0; p < 2; ++p) {
    const int n  = p * 32 + (tid >> 3);
    const int kc = (tid & 7) * 4;
    const float4 v = *(const float4*)(W + (long)n * ldw + k + kc);
    sB[(kc + 0) * BSTR + n] = (__bf16)v.x;
    sB[(kc + 1) * BSTR + n] = (__bf16)v.y;
    sB[(kc + 2) * BSTR + n] = (__bf16)v.z;
    sB[(kc + 3) * BSTR + n] = (__bf16)v.w;
  }
}

// ---- tiled GEMM accumulate: acc += A[m0:,:K] @ W[:64,:K]^T -----------------
// W must already be offset to the first of the 64 output rows.
DEVI void gemm_acc(v8f acc[2][2], const float* A, long lda,
                   const float* W, long ldw, int K, int m0,
                   __bf16* sA, __bf16* sB) {
  const int tid  = threadIdx.x;
  const int lane = tid & 31;
  const int wid  = tid >> 5;
  const int wm   = (wid >> 1) * 32;   // 0..96
  const int wn   = (wid & 1) * 32;    // 0 or 32

  for (int k = 0; k < K; k += TK) {
    __syncthreads();
    stage_tiles(A, lda, W, ldw, m0, k, tid, sA, sB);
    if (k + TK < K) {  // prefetch next K-slab (global_prefetch_b8)
      __builtin_prefetch((const void*)(A + (long)(m0 + (tid >> 3)) * lda + k + TK), 0, 1);
      __builtin_prefetch((const void*)(W + (long)(tid >> 3) * ldw + k + TK), 0, 1);
    }
    __syncthreads();

    const v16bf a0 = load_afrag(sA, wm,      lane);
    const v16bf a1 = load_afrag(sA, wm + 16, lane);
    const v16bf b0 = load_bfrag(sB, wn,      lane);
    const v16bf b1 = load_bfrag(sB, wn + 16, lane);

    acc[0][0] = __builtin_amdgcn_wmma_f32_16x16x32_bf16(false, a0, false, b0, (short)0, acc[0][0], false, false);
    acc[0][1] = __builtin_amdgcn_wmma_f32_16x16x32_bf16(false, a0, false, b1, (short)0, acc[0][1], false, false);
    acc[1][0] = __builtin_amdgcn_wmma_f32_16x16x32_bf16(false, a1, false, b0, (short)0, acc[1][0], false, false);
    acc[1][1] = __builtin_amdgcn_wmma_f32_16x16x32_bf16(false, a1, false, b1, (short)0, acc[1][1], false, false);
  }
}

DEVI void zero_acc(v8f a[2][2]) {
  v8f z = {};
  a[0][0] = z; a[0][1] = z; a[1][0] = z; a[1][1] = z;
}

// ===========================================================================
// Kernel 1: meta_pre[512,1024] = input@Wih[:, :I]^T + main_h@Wih[:, I:]^T
//                                + meta_h@Whh^T + bias_hyper
__global__ __launch_bounds__(256) void k_meta_pre(
    const float* __restrict__ input, const float* __restrict__ main_h,
    const float* __restrict__ meta_h, const float* __restrict__ w_ih,
    const float* __restrict__ w_hh, const float* __restrict__ bias_hyper,
    float* __restrict__ pre_out) {
  __shared__ __bf16 sA[LDSA];
  __shared__ __bf16 sB[LDSB];
  const int m0 = blockIdx.x * TM;
  const int n0 = blockIdx.y * TN;

  v8f acc[2][2]; zero_acc(acc);
  gemm_acc(acc, input,  Iw,  w_ih + (long)n0 * (Iw + Hw),      Iw + Hw, Iw,  m0, sA, sB);
  gemm_acc(acc, main_h, Hw,  w_ih + (long)n0 * (Iw + Hw) + Iw, Iw + Hw, Hw,  m0, sA, sB);
  gemm_acc(acc, meta_h, HHw, w_hh + (long)n0 * HHw,            HHw,     HHw, m0, sA, sB);

  const int lane = threadIdx.x & 31, wid = threadIdx.x >> 5;
  const int wm = (wid >> 1) * 32, wn = (wid & 1) * 32;
#pragma unroll
  for (int mi = 0; mi < 2; ++mi)
#pragma unroll
    for (int ni = 0; ni < 2; ++ni)
#pragma unroll
      for (int r = 0; r < 8; ++r) {
        const int row = m0 + wm + mi * 16 + r + ((lane >> 4) << 3);
        const int col = n0 + wn + ni * 16 + (lane & 15);
        pre_out[(long)row * 1024 + col] = acc[mi][ni][r] + bias_hyper[col];
      }
}

// Kernel 2: meta LSTM gates -> meta_h_new, meta_c_new (into d_out)
__global__ __launch_bounds__(256) void k_meta_gates(
    const float* __restrict__ pre, const float* __restrict__ meta_c,
    float* __restrict__ out_h, float* __restrict__ out_c) {
  const int idx = blockIdx.x * 256 + threadIdx.x;  // 0 .. 512*256-1
  const int b = idx >> 8, j = idx & 255;
  const float* p = pre + (long)b * 1024;
  const float mi_ = p[j], mf = p[j + 256], mg = p[j + 512], mo = p[j + 768];
  const float c = sigm(mf) * meta_c[idx] + sigm(mi_) * tanhf(mg);
  out_c[idx] = c;
  out_h[idx] = sigm(mo) * tanhf(c);
}

// Kernel 3: zbuf[zs][512][64] = meta_h_new @ Wz^T (+ bias)
__global__ __launch_bounds__(256) void k_z(
    const float* __restrict__ meta_h_new, const float* __restrict__ w_hzi,
    const float* __restrict__ w_hzH, const float* __restrict__ w_hzb,
    const float* __restrict__ bias_i, const float* __restrict__ bias_H,
    float* __restrict__ zbuf) {
  __shared__ __bf16 sA[LDSA];
  __shared__ __bf16 sB[LDSB];
  const int m0 = blockIdx.x * TM;
  const int zs = blockIdx.y;
  const float* W    = (zs == 0) ? w_hzi  : (zs == 1) ? w_hzH  : w_hzb;
  const float* bias = (zs == 0) ? bias_i : (zs == 1) ? bias_H : nullptr;

  v8f acc[2][2]; zero_acc(acc);
  gemm_acc(acc, meta_h_new, HHw, W, HHw, HHw, m0, sA, sB);

  const int lane = threadIdx.x & 31, wid = threadIdx.x >> 5;
  const int wm = (wid >> 1) * 32, wn = (wid & 1) * 32;
  float* dst = zbuf + (long)zs * Bn * Ew;
#pragma unroll
  for (int mi = 0; mi < 2; ++mi)
#pragma unroll
    for (int ni = 0; ni < 2; ++ni)
#pragma unroll
      for (int r = 0; r < 8; ++r) {
        const int row = m0 + wm + mi * 16 + r + ((lane >> 4) << 3);
        const int col = wn + ni * 16 + (lane & 15);
        float v = acc[mi][ni][r];
        if (bias) v += bias[col];
        dst[(long)row * Ew + col] = v;
      }
}

// Kernel 4: fused main pre + LSTM gates.
// WG handles rows m0..m0+127, H-columns h0..h0+63 across all 4 gates;
// pre fragments stay in registers, nonlinearity fused, writes h_new/c_new.
__global__ __launch_bounds__(256) void k_main(
    const float* __restrict__ input, const float* __restrict__ main_h,
    const float* __restrict__ main_c, const float* __restrict__ w_iH_all,
    const float* __restrict__ w_HH_all, const int* __restrict__ task_ptr,
    const float* __restrict__ w_dzi, const float* __restrict__ w_dzH,
    const float* __restrict__ w_bz, const float* __restrict__ bias,
    const float* __restrict__ zbuf,
    float* __restrict__ out_h, float* __restrict__ out_c) {
  __shared__ __bf16 sA[LDSA];
  __shared__ __bf16 sB[LDSB];
  const long task = (long)task_ptr[0];
  const float* w_iH = w_iH_all + task * (long)(4 * Hw) * Iw;
  const float* w_HH = w_HH_all + task * (long)(4 * Hw) * Hw;
  const float* zi = zbuf;
  const float* zH = zbuf + (long)Bn * Ew;
  const float* zb = zbuf + 2L * Bn * Ew;

  const int m0 = blockIdx.x * TM;
  const int h0 = blockIdx.y * TN;

  v8f pre[4][2][2];
  v8f accA[2][2], accB[2][2];

  for (int g = 0; g < 4; ++g) {
    const long cg = (long)g * Hw + h0;  // row offset into [4H, *] weights

    zero_acc(accA);
    gemm_acc(accA, input, Iw, w_iH + cg * Iw, Iw, Iw, m0, sA, sB);
    zero_acc(accB);
    gemm_acc(accB, zi, Ew, w_dzi + cg * Ew, Ew, Ew, m0, sA, sB);
#pragma unroll
    for (int mi = 0; mi < 2; ++mi)
#pragma unroll
      for (int ni = 0; ni < 2; ++ni)
        pre[g][mi][ni] = accA[mi][ni] * accB[mi][ni];

    zero_acc(accA);
    gemm_acc(accA, main_h, Hw, w_HH + cg * Hw, Hw, Hw, m0, sA, sB);
    zero_acc(accB);
    gemm_acc(accB, zH, Ew, w_dzH + cg * Ew, Ew, Ew, m0, sA, sB);
#pragma unroll
    for (int mi = 0; mi < 2; ++mi)
#pragma unroll
      for (int ni = 0; ni < 2; ++ni)
        pre[g][mi][ni] += accA[mi][ni] * accB[mi][ni];

    zero_acc(accB);
    gemm_acc(accB, zb, Ew, w_bz + cg * Ew, Ew, Ew, m0, sA, sB);
#pragma unroll
    for (int mi = 0; mi < 2; ++mi)
#pragma unroll
      for (int ni = 0; ni < 2; ++ni)
        pre[g][mi][ni] += accB[mi][ni];
  }

  const int lane = threadIdx.x & 31, wid = threadIdx.x >> 5;
  const int wm = (wid >> 1) * 32, wn = (wid & 1) * 32;
#pragma unroll
  for (int mi = 0; mi < 2; ++mi)
#pragma unroll
    for (int ni = 0; ni < 2; ++ni)
#pragma unroll
      for (int r = 0; r < 8; ++r) {
        const int row = m0 + wm + mi * 16 + r + ((lane >> 4) << 3);
        const int col = h0 + wn + ni * 16 + (lane & 15);
        const float gi = sigm(pre[0][mi][ni][r] + bias[col]);
        const float gf = sigm(pre[1][mi][ni][r] + bias[Hw + col]);
        const float gg = tanhf(pre[2][mi][ni][r] + bias[2 * Hw + col]);
        const float go = sigm(pre[3][mi][ni][r] + bias[3 * Hw + col]);
        const long o = (long)row * Hw + col;
        const float cn = gf * main_c[o] + gi * gg;
        out_c[o] = cn;
        out_h[o] = go * tanhf(cn);
      }
}

// ===========================================================================
extern "C" void kernel_launch(void* const* d_in, const int* in_sizes, int n_in,
                              void* d_out, int out_size, void* d_ws, size_t ws_size,
                              hipStream_t stream) {
  const float* input      = (const float*)d_in[0];
  const float* main_h     = (const float*)d_in[1];
  const float* main_c     = (const float*)d_in[2];
  const float* meta_h     = (const float*)d_in[3];
  const float* meta_c     = (const float*)d_in[4];
  const int*   task_index = (const int*)  d_in[5];
  const float* weight_iH  = (const float*)d_in[6];
  const float* weight_HH  = (const float*)d_in[7];
  const float* weight_ih  = (const float*)d_in[8];
  const float* weight_hh  = (const float*)d_in[9];
  const float* weight_hzi = (const float*)d_in[10];
  const float* weight_hzH = (const float*)d_in[11];
  const float* weight_hzb = (const float*)d_in[12];
  const float* weight_dzi = (const float*)d_in[13];
  const float* weight_dzH = (const float*)d_in[14];
  const float* weight_bz  = (const float*)d_in[15];
  const float* bias_i     = (const float*)d_in[16];
  const float* bias_H     = (const float*)d_in[17];
  const float* bias       = (const float*)d_in[18];
  const float* bias_hyper = (const float*)d_in[19];

  float* ws       = (float*)d_ws;
  float* meta_pre = ws;                       // 512*1024 f32
  float* zbuf     = ws + (long)Bn * 1024;     // 3*512*64 f32

  float* out        = (float*)d_out;
  float* out_main_h = out;
  float* out_main_c = out + (long)Bn * Hw;
  float* out_meta_h = out + 2L * Bn * Hw;
  float* out_meta_c = out_meta_h + (long)Bn * HHw;

  k_meta_pre<<<dim3(Bn / TM, 1024 / TN), 256, 0, stream>>>(
      input, main_h, meta_h, weight_ih, weight_hh, bias_hyper, meta_pre);

  k_meta_gates<<<dim3((Bn * HHw) / 256), 256, 0, stream>>>(
      meta_pre, meta_c, out_meta_h, out_meta_c);

  k_z<<<dim3(Bn / TM, 3), 256, 0, stream>>>(
      out_meta_h, weight_hzi, weight_hzH, weight_hzb, bias_i, bias_H, zbuf);

  k_main<<<dim3(Bn / TM, Hw / TN), 256, 0, stream>>>(
      input, main_h, main_c, weight_iH, weight_HH, task_index,
      weight_dzi, weight_dzH, weight_bz, bias, zbuf, out_main_h, out_main_c);
}
